// RGCN_LP_40037685133763
// MI455X (gfx1250) — matrix operand order repaired and checked by reading, not compiled, not run on hardware
//
#include <hip/hip_runtime.h>
#include <hip/hip_bf16.h>
#include <stdint.h>

// Problem constants (match reference)
#define NN      50000
#define INC     128
#define HID     256
#define RREL    8
#define NBASES  30
#define EE      800000
#define BB      200000

typedef __attribute__((ext_vector_type(16))) __bf16 v16bf;
typedef __attribute__((ext_vector_type(8)))  float  v8f;

__device__ __forceinline__ unsigned short f2bf(float f) {
  union { float f; unsigned int u; } x; x.f = f;
  unsigned int u = x.u;
  unsigned int r = (u + 0x7FFFu + ((u >> 16) & 1u)) >> 16;  // RNE
  return (unsigned short)r;
}
__device__ __forceinline__ float bf2f(unsigned short h) {
  union { unsigned int u; float f; } x; x.u = ((unsigned int)h) << 16;
  return x.f;
}

// ---------------- utility ----------------
__global__ void zero_f32_kernel(float* __restrict__ p, long long n) {
  for (long long i = blockIdx.x * (long long)blockDim.x + threadIdx.x; i < n;
       i += (long long)gridDim.x * blockDim.x)
    p[i] = 0.f;
}

// ---------------- weight preparation ----------------
__global__ void build_wt_rgcn_kernel(const float* __restrict__ basis,
                                     const float* __restrict__ comp,
                                     const float* __restrict__ root,
                                     unsigned short* __restrict__ Wt,
                                     int Cin, int Cout) {
  const int K = RREL * Cin + Cin;
  const long long total = (long long)Cout * K;
  for (long long idx = blockIdx.x * (long long)blockDim.x + threadIdx.x; idx < total;
       idx += (long long)gridDim.x * blockDim.x) {
    int o = (int)(idx / K);
    int k = (int)(idx % K);
    float v;
    if (k < RREL * Cin) {
      int r = k / Cin, i = k % Cin;
      v = 0.f;
      #pragma unroll 5
      for (int b = 0; b < NBASES; ++b)
        v += comp[r * NBASES + b] * basis[((size_t)b * Cin + i) * Cout + o];
    } else {
      int i = k - RREL * Cin;
      v = root[(size_t)i * Cout + o];
    }
    Wt[(size_t)o * K + k] = f2bf(v);
  }
}

// W [K,Nc] fp32 row-major -> Wt [Nc,K] bf16
__global__ void transpose_bf16_kernel(const float* __restrict__ W,
                                      unsigned short* __restrict__ Wt,
                                      int K, int Nc) {
  const long long total = (long long)K * Nc;
  for (long long idx = blockIdx.x * (long long)blockDim.x + threadIdx.x; idx < total;
       idx += (long long)gridDim.x * blockDim.x) {
    int k = (int)(idx / Nc);
    int n = (int)(idx % Nc);
    Wt[(size_t)n * K + k] = f2bf(W[idx]);
  }
}

// ---------------- edge aggregation (scatter, C is compile-time pow2) ----------------
template <int C>
__global__ void scatter_f32_kernel(const float* __restrict__ feat,
                                   const int* __restrict__ src,
                                   const int* __restrict__ dst,
                                   const int* __restrict__ et,
                                   float* __restrict__ accum,
                                   float* __restrict__ cnt, long long total) {
  for (long long idx = blockIdx.x * (long long)blockDim.x + threadIdx.x; idx < total;
       idx += (long long)gridDim.x * blockDim.x) {
    long long e = idx / C;            // shift (C constexpr pow2)
    int c = (int)(idx % C);
    int s = src[e], d = dst[e], r = et[e];
    atomicAdd(&accum[((size_t)d * RREL + r) * C + c], feat[(size_t)s * C + c]);
    if (c == 0) atomicAdd(&cnt[(size_t)d * RREL + r], 1.0f);
  }
}

template <int C>
__global__ void scatter_bf16_kernel(const unsigned short* __restrict__ feat,
                                    const int* __restrict__ src,
                                    const int* __restrict__ dst,
                                    const int* __restrict__ et,
                                    float* __restrict__ accum,
                                    float* __restrict__ cnt, long long total) {
  for (long long idx = blockIdx.x * (long long)blockDim.x + threadIdx.x; idx < total;
       idx += (long long)gridDim.x * blockDim.x) {
    long long e = idx / C;
    int c = (int)(idx % C);
    int s = src[e], d = dst[e], r = et[e];
    atomicAdd(&accum[((size_t)d * RREL + r) * C + c], bf2f(feat[(size_t)s * C + c]));
    if (c == 0) atomicAdd(&cnt[(size_t)d * RREL + r], 1.0f);
  }
}

// ---------------- pack A = [mean(n,r,:) r=0..7 || self(n,:)] bf16 ----------------
template <int C>
__global__ void pack_mean_f32self_kernel(const float* __restrict__ accum,
                                         const float* __restrict__ cnt,
                                         const float* __restrict__ self,
                                         unsigned short* __restrict__ A, int Nn) {
  constexpr int K = (RREL + 1) * C;
  const long long total = (long long)Nn * K;
  for (long long idx = blockIdx.x * (long long)blockDim.x + threadIdx.x; idx < total;
       idx += (long long)gridDim.x * blockDim.x) {
    int n = (int)(idx / K);           // div by constant -> mul/shift
    int k = (int)(idx % K);
    float v;
    if (k < RREL * C) {
      int r = k / C, i = k % C;
      float cc = cnt[(size_t)n * RREL + r];
      v = accum[((size_t)n * RREL + r) * C + i] / fmaxf(cc, 1.0f);
    } else {
      v = self[(size_t)n * C + (k - RREL * C)];
    }
    A[(size_t)n * K + k] = f2bf(v);
  }
}

template <int C>
__global__ void pack_mean_bf16self_kernel(const float* __restrict__ accum,
                                          const float* __restrict__ cnt,
                                          const unsigned short* __restrict__ self,
                                          unsigned short* __restrict__ A, int Nn) {
  constexpr int K = (RREL + 1) * C;
  const long long total = (long long)Nn * K;
  for (long long idx = blockIdx.x * (long long)blockDim.x + threadIdx.x; idx < total;
       idx += (long long)gridDim.x * blockDim.x) {
    int n = (int)(idx / K);
    int k = (int)(idx % K);
    float v;
    if (k < RREL * C) {
      int r = k / C, i = k % C;
      float cc = cnt[(size_t)n * RREL + r];
      v = accum[((size_t)n * RREL + r) * C + i] / fmaxf(cc, 1.0f);
    } else {
      v = bf2f(self[(size_t)n * C + (k - RREL * C)]);
    }
    A[(size_t)n * K + k] = f2bf(v);
  }
}

// ---------------- WMMA GEMM: Y[M,Nc] = A[M,K]bf16 @ Wt[Nc,K]bf16^T + bias --------
// Wave computes 16x64; 2-deep software pipeline (requires K % 64 == 0, M % 16 == 0,
// Nc % 64 == 0). A/B fragments: contiguous K per lane half (k = 16*hi + s).
__global__ void __launch_bounds__(256) gemm_bf16_wmma_kernel(
    const unsigned short* __restrict__ A, const unsigned short* __restrict__ Wt,
    const float* __restrict__ bias, float* __restrict__ Y,
    int M, int K, int Nc) {
  const int lane = threadIdx.x & 31;
  const int wave = threadIdx.x >> 5;
  const int hi = lane >> 4;
  const int lr = lane & 15;
  const int nG = Nc >> 6;
  const int total = (M >> 4) * nG;
  const int tile = blockIdx.x * 8 + wave;
  if (tile >= total) return;
  const int mT = tile / nG;
  const int gT = tile % nG;

  const unsigned short* aRow  = A  + (size_t)(mT * 16 + lr) * K;
  const unsigned short* bRow0 = Wt + (size_t)(gT * 64 + lr) * K;
  const size_t bStride = (size_t)16 * K;

  union Frag { uint4 q[2]; v16bf v; };
  Frag fa[2], fb[2][4];
  v8f acc[4];
  #pragma unroll
  for (int t = 0; t < 4; ++t) acc[t] = (v8f){0.f, 0.f, 0.f, 0.f, 0.f, 0.f, 0.f, 0.f};

  auto loadstep = [&](int buf, int k0) {
    // A 16x32 bf16: lane (hi,lr) holds 16 contiguous K at hi*16
    fa[buf].q[0] = *reinterpret_cast<const uint4*>(aRow + k0 + hi * 16);
    fa[buf].q[1] = *reinterpret_cast<const uint4*>(aRow + k0 + hi * 16 + 8);
    #pragma unroll
    for (int t = 0; t < 4; ++t) {
      // B 32x16 bf16: lane holds column n=lr, 16 contiguous K at hi*16
      const unsigned short* bp = bRow0 + bStride * t + k0 + hi * 16;
      fb[buf][t].q[0] = *reinterpret_cast<const uint4*>(bp);
      fb[buf][t].q[1] = *reinterpret_cast<const uint4*>(bp + 8);
    }
  };
  auto mmstep = [&](int buf) {
    #pragma unroll
    for (int t = 0; t < 4; ++t)
      acc[t] = __builtin_amdgcn_wmma_f32_16x16x32_bf16(
          false, fa[buf].v, false, fb[buf][t].v, (short)0, acc[t], false, false);
  };

  loadstep(0, 0);
  for (int k0 = 0; k0 < K; k0 += 64) {      // K % 64 == 0 at every call site
    __builtin_prefetch(aRow + k0 + 128, 0, 3);
    loadstep(1, k0 + 32);                   // stage k+32 in flight during mm(k)
    mmstep(0);
    if (k0 + 64 < K) loadstep(0, k0 + 64);  // stage k+64 in flight during mm(k+32)
    mmstep(1);
  }

  // D: lane (hi,lr) holds rows m = hi*8 + v, col n = lr
  float* yBase = Y + (size_t)(mT * 16 + hi * 8) * Nc;
  const int nBase = gT * 64 + lr;
  #pragma unroll
  for (int t = 0; t < 4; ++t) {
    const int n = nBase + t * 16;
    const float bb = bias[n];
    #pragma unroll
    for (int v = 0; v < 8; ++v)
      yBase[(size_t)v * Nc + n] = acc[t][v] + bb;
  }
}

// ---------------- BatchNorm (training-mode, biased var), coalesced ----------------
template <int C>
__global__ void __launch_bounds__(256) bn_partial_kernel(const float* __restrict__ Y,
                                                         float* __restrict__ sums,
                                                         int M) {
  const int tid = threadIdx.x;
  if constexpr (C <= 256) {
    constexpr int RPB = 256 / C;            // rows covered per block step
    const int ro = tid / C;
    const int c = tid % C;
    float s = 0.f, q = 0.f;
    for (int r = blockIdx.x * RPB + ro; r < M; r += gridDim.x * RPB) {
      float v = Y[(size_t)r * C + c];       // consecutive lanes -> consecutive cols
      s += v; q += v * v;
    }
    atomicAdd(&sums[c], s);
    atomicAdd(&sums[C + c], q);
  } else {
    constexpr int CPT = C / 256;
    float s[CPT] = {}, q[CPT] = {};
    for (int r = blockIdx.x; r < M; r += gridDim.x) {
      const float* row = Y + (size_t)r * C;
      #pragma unroll
      for (int j = 0; j < CPT; ++j) {
        float v = row[tid + j * 256];
        s[j] += v; q[j] += v * v;
      }
    }
    #pragma unroll
    for (int j = 0; j < CPT; ++j) {
      atomicAdd(&sums[tid + j * 256], s[j]);
      atomicAdd(&sums[C + tid + j * 256], q[j]);
    }
  }
}

__global__ void bn_finalize_kernel(const float* __restrict__ sums,
                                   float* __restrict__ stats, int M, int C) {
  int c = blockIdx.x * blockDim.x + threadIdx.x;
  if (c >= C) return;
  float mean = sums[c] / (float)M;
  float var = sums[C + c] / (float)M - mean * mean;
  stats[c] = mean;
  stats[C + c] = rsqrtf(var + 1e-5f);
}

template <int C>
__global__ void bn_apply_relu_kernel(const float* __restrict__ Y,
                                     const float* __restrict__ stats,
                                     const float* __restrict__ g,
                                     const float* __restrict__ b,
                                     unsigned short* __restrict__ out,
                                     long long total) {
  for (long long i = blockIdx.x * (long long)blockDim.x + threadIdx.x; i < total;
       i += (long long)gridDim.x * blockDim.x) {
    int c = (int)(i & (long long)(C - 1));  // C pow2
    float v = (Y[i] - stats[c]) * stats[C + c] * g[c] + b[c];
    out[i] = f2bf(v > 0.f ? v : 0.f);
  }
}

// ---------------- decode: gather pair + AvgPool1d(2,2) ----------------
__global__ void pool_kernel(const unsigned short* __restrict__ z,
                            const int* __restrict__ eli,   // [2,B]
                            float* __restrict__ pooled,    // [B, HID] fp32
                            int Bn) {
  const long long total = (long long)Bn * HID;
  for (long long idx = blockIdx.x * (long long)blockDim.x + threadIdx.x; idx < total;
       idx += (long long)gridDim.x * blockDim.x) {
    int b = (int)(idx / HID);
    int c = (int)(idx % HID);
    int node, cc;
    if (c < HID / 2) { node = eli[b];      cc = 2 * c; }
    else             { node = eli[Bn + b]; cc = 2 * (c - HID / 2); }
    float v = 0.5f * (bf2f(z[(size_t)node * HID + cc]) +
                      bf2f(z[(size_t)node * HID + cc + 1]));
    pooled[idx] = v;
  }
}

// ---------------- final fc5: [B,64] @ [64,1] + b ----------------
__global__ void final_fc_kernel(const unsigned short* __restrict__ h,
                                const float* __restrict__ w,
                                const float* __restrict__ b,
                                float* __restrict__ out, int Bn) {
  int i = blockIdx.x * blockDim.x + threadIdx.x;
  if (i >= Bn) return;
  float s = b[0];
  #pragma unroll
  for (int k = 0; k < 64; ++k) s += bf2f(h[(size_t)i * 64 + k]) * w[k];
  out[i] = s;
}

// ---------------- host helpers ----------------
static inline unsigned int nblocks(long long total, int bs) {
  long long b = (total + bs - 1) / bs;
  if (b > (1LL << 20)) b = 1LL << 20;
  return (unsigned int)b;
}

template <int C>
static void run_bn(const float* Y, const float* g, const float* b,
                   unsigned short* o, int M, float* sums, float* stats,
                   hipStream_t stream) {
  zero_f32_kernel<<<nblocks(2 * C, 256), 256, 0, stream>>>(sums, 2 * C);
  bn_partial_kernel<C><<<768, 256, 0, stream>>>(Y, sums, M);
  bn_finalize_kernel<<<nblocks(C, 256), 256, 0, stream>>>(sums, stats, M, C);
  long long t = (long long)M * C;
  bn_apply_relu_kernel<C><<<nblocks(t, 256), 256, 0, stream>>>(Y, stats, g, b, o, t);
}

extern "C" void kernel_launch(void* const* d_in, const int* in_sizes, int n_in,
                              void* d_out, int out_size, void* d_ws, size_t ws_size,
                              hipStream_t stream) {
  const float* x      = (const float*)d_in[0];
  const int*   ei     = (const int*)d_in[1];   // [2,E]: src=ei, dst=ei+E
  const int*   et     = (const int*)d_in[2];
  const int*   eli    = (const int*)d_in[3];   // [2,B]
  const float* basis1 = (const float*)d_in[4];
  const float* comp1  = (const float*)d_in[5];
  const float* root1  = (const float*)d_in[6];
  const float* bias1  = (const float*)d_in[7];
  const float* bn1g   = (const float*)d_in[8];
  const float* bn1b   = (const float*)d_in[9];
  const float* basis2 = (const float*)d_in[10];
  const float* comp2  = (const float*)d_in[11];
  const float* root2  = (const float*)d_in[12];
  const float* bias2  = (const float*)d_in[13];
  const float* bn2g   = (const float*)d_in[14];
  const float* bn2b   = (const float*)d_in[15];
  const float* bnpg   = (const float*)d_in[16];
  const float* bnpb   = (const float*)d_in[17];
  const float* fc1w   = (const float*)d_in[18];
  const float* fc1b   = (const float*)d_in[19];
  const float* bn3g   = (const float*)d_in[20];
  const float* bn3b   = (const float*)d_in[21];
  const float* fc2w   = (const float*)d_in[22];
  const float* fc2b   = (const float*)d_in[23];
  const float* bn4g   = (const float*)d_in[24];
  const float* bn4b   = (const float*)d_in[25];
  const float* fc3w   = (const float*)d_in[26];
  const float* fc3b   = (const float*)d_in[27];
  const float* bn5g   = (const float*)d_in[28];
  const float* bn5b   = (const float*)d_in[29];
  const float* fc4w   = (const float*)d_in[30];
  const float* fc4b   = (const float*)d_in[31];
  const float* bn6g   = (const float*)d_in[32];
  const float* bn6b   = (const float*)d_in[33];
  const float* fc5w   = (const float*)d_in[34];
  const float* fc5b   = (const float*)d_in[35];
  float* out = (float*)d_out;

  // --- workspace carve (256B aligned) ---
  char* ws = (char*)d_ws;
  size_t off = 0;
  auto carve = [&](size_t bytes) -> char* {
    char* p = ws + off;
    off = (off + bytes + 255) & ~(size_t)255;
    return p;
  };
  float* accum = (float*)carve((size_t)102400000 * 4);  // N*R*256 fp32, reused as GEMM out (B*512)
  float* Ybuf  = accum;
  float* cnt   = (float*)carve((size_t)NN * RREL * 4);
  unsigned short* Abuf = (unsigned short*)carve((size_t)NN * 2304 * 2);  // also h1 (B*512)
  unsigned short* h1   = Abuf;
  unsigned short* z1   = (unsigned short*)carve((size_t)NN * HID * 2);
  unsigned short* z2   = (unsigned short*)carve((size_t)NN * HID * 2);
  unsigned short* h0   = (unsigned short*)carve((size_t)BB * 256 * 2);
  unsigned short* h2   = (unsigned short*)carve((size_t)BB * 256 * 2);
  unsigned short* h3   = (unsigned short*)carve((size_t)BB * 128 * 2);
  unsigned short* h4   = (unsigned short*)carve((size_t)BB * 64 * 2);
  unsigned short* Wt1  = (unsigned short*)carve((size_t)256 * 1152 * 2);
  unsigned short* Wt2  = (unsigned short*)carve((size_t)256 * 2304 * 2);
  unsigned short* f1t  = (unsigned short*)carve((size_t)512 * 256 * 2);
  unsigned short* f2t  = (unsigned short*)carve((size_t)256 * 512 * 2);
  unsigned short* f3t  = (unsigned short*)carve((size_t)128 * 256 * 2);
  unsigned short* f4t  = (unsigned short*)carve((size_t)64 * 128 * 2);
  float* sums  = (float*)carve((size_t)1024 * 4);
  float* stats = (float*)carve((size_t)1024 * 4);

  auto run_gemm = [&](const unsigned short* A, const unsigned short* Wt,
                      const float* bias, float* Y, int M, int K, int Nc) {
    int tiles = (M >> 4) * (Nc >> 6);
    gemm_bf16_wmma_kernel<<<(tiles + 7) / 8, 256, 0, stream>>>(A, Wt, bias, Y, M, K, Nc);
  };

  const int* src = ei;
  const int* dst = ei + EE;

  // --- weight prep ---
  build_wt_rgcn_kernel<<<nblocks((long long)256 * 1152, 256), 256, 0, stream>>>(
      basis1, comp1, root1, Wt1, INC, HID);
  build_wt_rgcn_kernel<<<nblocks((long long)256 * 2304, 256), 256, 0, stream>>>(
      basis2, comp2, root2, Wt2, HID, HID);
  transpose_bf16_kernel<<<nblocks(256 * 512, 256), 256, 0, stream>>>(fc1w, f1t, 256, 512);
  transpose_bf16_kernel<<<nblocks(512 * 256, 256), 256, 0, stream>>>(fc2w, f2t, 512, 256);
  transpose_bf16_kernel<<<nblocks(256 * 128, 256), 256, 0, stream>>>(fc3w, f3t, 256, 128);
  transpose_bf16_kernel<<<nblocks(128 * 64, 256), 256, 0, stream>>>(fc4w, f4t, 128, 64);

  // --- RGCN layer 1 ---
  zero_f32_kernel<<<nblocks((long long)NN * RREL * INC, 256), 256, 0, stream>>>(
      accum, (long long)NN * RREL * INC);
  zero_f32_kernel<<<nblocks((long long)NN * RREL, 256), 256, 0, stream>>>(
      cnt, (long long)NN * RREL);
  scatter_f32_kernel<INC><<<nblocks((long long)EE * INC, 256), 256, 0, stream>>>(
      x, src, dst, et, accum, cnt, (long long)EE * INC);
  pack_mean_f32self_kernel<INC><<<nblocks((long long)NN * 1152, 256), 256, 0, stream>>>(
      accum, cnt, x, Abuf, NN);
  run_gemm(Abuf, Wt1, bias1, Ybuf, NN, 1152, HID);
  run_bn<HID>(Ybuf, bn1g, bn1b, z1, NN, sums, stats, stream);

  // --- RGCN layer 2 ---
  zero_f32_kernel<<<nblocks((long long)NN * RREL * HID, 256), 256, 0, stream>>>(
      accum, (long long)NN * RREL * HID);
  zero_f32_kernel<<<nblocks((long long)NN * RREL, 256), 256, 0, stream>>>(
      cnt, (long long)NN * RREL);
  scatter_bf16_kernel<HID><<<nblocks((long long)EE * HID, 256), 256, 0, stream>>>(
      z1, src, dst, et, accum, cnt, (long long)EE * HID);
  pack_mean_bf16self_kernel<HID><<<nblocks((long long)NN * 2304, 256), 256, 0, stream>>>(
      accum, cnt, z1, Abuf, NN);
  run_gemm(Abuf, Wt2, bias2, Ybuf, NN, 2304, HID);
  run_bn<HID>(Ybuf, bn2g, bn2b, z2, NN, sums, stats, stream);

  // --- decode: gather + pool + BN ---
  pool_kernel<<<nblocks((long long)BB * HID, 256), 256, 0, stream>>>(z2, eli, Ybuf, BB);
  run_bn<256>(Ybuf, bnpg, bnpb, h0, BB, sums, stats, stream);

  // --- MLP ---
  run_gemm(h0, f1t, fc1b, Ybuf, BB, 256, 512);
  run_bn<512>(Ybuf, bn3g, bn3b, h1, BB, sums, stats, stream);
  run_gemm(h1, f2t, fc2b, Ybuf, BB, 512, 256);
  run_bn<256>(Ybuf, bn4g, bn4b, h2, BB, sums, stats, stream);
  run_gemm(h2, f3t, fc3b, Ybuf, BB, 256, 128);
  run_bn<128>(Ybuf, bn5g, bn5b, h3, BB, sums, stats, stream);
  run_gemm(h3, f4t, fc4b, Ybuf, BB, 128, 64);
  run_bn<64>(Ybuf, bn6g, bn6b, h4, BB, sums, stats, stream);
  final_fc_kernel<<<(BB + 255) / 256, 256, 0, stream>>>(h4, fc5w, fc5b, out, BB);

  (void)in_sizes; (void)n_in; (void)out_size; (void)ws_size;
}
